// myGPT2Block_68934225101634
// MI455X (gfx1250) — matrix operand order repaired and checked by reading, not compiled, not run on hardware
//
#include <hip/hip_runtime.h>
#include <math.h>

// ---------------------------------------------------------------------------
// GPT-2 block for MI455X (gfx1250): bf16 WMMA GEMMs + flash attention.
// All GEMMs: pre-transposed bf16 weights [N][K], 128x128 block tile,
// K-tile 64, double-buffered LDS, 16 WMMAs per K-tile per wave.
// ---------------------------------------------------------------------------

typedef __bf16 bf16_t;
typedef __attribute__((ext_vector_type(16))) __bf16 v16bf;
typedef __attribute__((ext_vector_type(8)))  __bf16 v8bf;
typedef __attribute__((ext_vector_type(8)))  float  v8f;

union FragAB { v16bf v; v8bf h[2]; };

#define BB   2
#define TT   2048
#define DDIM 1024
#define NH   16
#define HDIM 64
#define FFD  4096

// ---------------------------------------------------------------------------
// fp32 [K][N] -> bf16 [N][K] transposed convert (weights, once per launch)
// ---------------------------------------------------------------------------
__global__ void cvt_transpose_bf16(const float* __restrict__ src,
                                   bf16_t* __restrict__ dst, int K, int N) {
    int i = blockIdx.x * blockDim.x + threadIdx.x;
    const int total = K * N;
    const int stride = gridDim.x * blockDim.x;
    for (; i < total; i += stride) {
        const int n = i / K, k = i % K;
        dst[i] = (bf16_t)src[(size_t)k * N + n];
    }
}

// ---------------------------------------------------------------------------
// LayerNorm over last dim (D=1024), one block per row, bf16 output.
// ---------------------------------------------------------------------------
__global__ void __launch_bounds__(256)
layernorm_bf16(const float* __restrict__ x, const float* __restrict__ w,
               const float* __restrict__ b, bf16_t* __restrict__ out, int D) {
    const int row = blockIdx.x;
    const float* xr = x + (size_t)row * D;
    float s = 0.f, s2 = 0.f;
    for (int c = threadIdx.x; c < D; c += blockDim.x) {
        float v = xr[c]; s += v; s2 += v * v;
    }
    for (int m = 16; m; m >>= 1) {
        s  += __shfl_xor(s,  m, 32);
        s2 += __shfl_xor(s2, m, 32);
    }
    __shared__ float rs[8], rs2[8];
    const int wave = threadIdx.x >> 5, lane = threadIdx.x & 31;
    if (lane == 0) { rs[wave] = s; rs2[wave] = s2; }
    __syncthreads();
    if (wave == 0) {
        s  = (lane < 8) ? rs[lane]  : 0.f;
        s2 = (lane < 8) ? rs2[lane] : 0.f;
        for (int m = 4; m; m >>= 1) {
            s  += __shfl_xor(s,  m, 32);
            s2 += __shfl_xor(s2, m, 32);
        }
        if (lane == 0) { rs[0] = s; rs2[0] = s2; }
    }
    __syncthreads();
    const float mu   = rs[0] / D;
    const float var  = rs2[0] / D - mu * mu;
    const float rinv = rsqrtf(var + 1e-5f);
    bf16_t* orow = out + (size_t)row * D;
    for (int c = threadIdx.x; c < D; c += blockDim.x)
        orow[c] = (bf16_t)((xr[c] - mu) * rinv * w[c] + b[c]);
}

// ---------------------------------------------------------------------------
// Tiled bf16 WMMA GEMM: C[M,N] = A[M,K] * BT[N,K]^T (+bias, gelu, residual).
// 256 threads = 8 waves (4 row-groups x 2 col-groups); wave computes 32x64.
// Double-buffered LDS; global loads issued before compute to overlap.
// VT: scatter bf16 output in [B,H,HD,T] layout (for attention V).
// ---------------------------------------------------------------------------
template<bool BIAS, bool GELU, bool RESID, bool OUTBF, bool VT>
__global__ void __launch_bounds__(256)
gemm_bf16_wmma(const bf16_t* __restrict__ A, const bf16_t* __restrict__ BT,
               const float* __restrict__ bias, const float* __restrict__ resid,
               void* __restrict__ Cout, int M, int N, int K) {
    __shared__ bf16_t As[2][128 * 64];   // [row][k]
    __shared__ bf16_t Bs[2][128 * 64];   // [n][k]
    const int tid  = threadIdx.x;
    const int lane = tid & 31, wave = tid >> 5;
    const int rowBase = blockIdx.y * 128;
    const int colBase = blockIdx.x * 128;
    const int r  = lane & 15;
    const int hi = lane >> 4;
    const int koffA = hi ? 8 : 0;    // A-frag: lanes>=16 hold K 8..15,24..31
    const int koffB = hi ? 8 : 0;    // B-frag: lanes>=16 hold K 8..15,24..31
    const int wrow = (wave >> 1) * 32;
    const int wcol = (wave & 1) * 64;

    v8f acc[8] = {};
    v8bf ra[4], rb[4];

    auto load_tile = [&](int k0) {
#pragma unroll
        for (int c = 0; c < 4; ++c) {
            const int chunk = tid * 4 + c;       // 0..1023
            const int rowi  = chunk >> 3;        // 0..127
            const int kc    = (chunk & 7) * 8;   // 0..56
            ra[c] = *(const v8bf*)(A  + (size_t)(rowBase + rowi) * K + k0 + kc);
            rb[c] = *(const v8bf*)(BT + (size_t)(colBase + rowi) * K + k0 + kc);
        }
    };
    auto store_tile = [&](int buf) {
#pragma unroll
        for (int c = 0; c < 4; ++c) {
            const int chunk = tid * 4 + c;
            const int rowi  = chunk >> 3;
            const int kc    = (chunk & 7) * 8;
            *(v8bf*)&As[buf][rowi * 64 + kc] = ra[c];
            *(v8bf*)&Bs[buf][rowi * 64 + kc] = rb[c];
        }
    };

    const int nkt = K >> 6;
    load_tile(0);
    store_tile(0);
    __syncthreads();

    for (int kt = 0; kt < nkt; ++kt) {
        const int buf = kt & 1;
        if (kt + 1 < nkt) load_tile((kt + 1) << 6);   // overlap with compute
#pragma unroll
        for (int ks = 0; ks < 2; ++ks) {
            const int kk = ks * 32;
            FragAB a0, a1;
            a0.h[0] = *(const v8bf*)&As[buf][(wrow + r) * 64 + kk + koffA];
            a0.h[1] = *(const v8bf*)&As[buf][(wrow + r) * 64 + kk + koffA + 16];
            a1.h[0] = *(const v8bf*)&As[buf][(wrow + 16 + r) * 64 + kk + koffA];
            a1.h[1] = *(const v8bf*)&As[buf][(wrow + 16 + r) * 64 + kk + koffA + 16];
#pragma unroll
            for (int f = 0; f < 4; ++f) {
                FragAB bfr;
                const bf16_t* bp = &Bs[buf][(wcol + f * 16 + r) * 64 + kk + koffB];
                bfr.h[0] = *(const v8bf*)bp;
                bfr.h[1] = *(const v8bf*)(bp + 16);
                acc[f] = __builtin_amdgcn_wmma_f32_16x16x32_bf16(
                    false, a0.v, false, bfr.v, (short)0, acc[f], false, false);
                acc[4 + f] = __builtin_amdgcn_wmma_f32_16x16x32_bf16(
                    false, a1.v, false, bfr.v, (short)0, acc[4 + f], false, false);
            }
        }
        if (kt + 1 < nkt) store_tile(buf ^ 1);
        __syncthreads();
    }

    // fused epilogue; C layout: element j of frag -> row j+8*hi, col lane&15
#pragma unroll
    for (int g = 0; g < 2; ++g) {
#pragma unroll
        for (int f = 0; f < 4; ++f) {
#pragma unroll
            for (int j = 0; j < 8; ++j) {
                const int m   = j + (hi << 3);
                const int row = rowBase + wrow + g * 16 + m;
                const int col = colBase + wcol + f * 16 + r;
                float v = acc[g * 4 + f][j];
                if (BIAS) v += bias[col];
                if (GELU) {
                    const float u = 0.7978845608028654f * (v + 0.044715f * v * v * v);
                    v = 0.5f * v * (1.0f + tanhf(u));
                }
                if (RESID) v += resid[(size_t)row * N + col];
                if (VT) {
                    // out[b][h][d][t] = v ; row = b*T + t, col = h*64 + d
                    const int bv = row >> 11, t = row & (TT - 1);
                    const int hv = col >> 6,  d = col & (HDIM - 1);
                    ((bf16_t*)Cout)[(size_t)((bv * NH + hv) * HDIM + d) * TT + t] = (bf16_t)v;
                } else if (OUTBF) {
                    ((bf16_t*)Cout)[(size_t)row * N + col] = (bf16_t)v;
                } else {
                    ((float*)Cout)[(size_t)row * N + col] = v;
                }
            }
        }
    }
}

// ---------------------------------------------------------------------------
// Flash attention (causal), bf16 WMMA. One wave owns a 16-row Q tile;
// streams 32-key K/V tiles through per-wave LDS. V is pre-transposed
// [B,H,HD,T] so staging is all contiguous b128 traffic. 4 waves / block.
// ---------------------------------------------------------------------------
__global__ void __launch_bounds__(128)
flash_attn_wmma(const bf16_t* __restrict__ qg, const bf16_t* __restrict__ kg,
                int ldqk, const bf16_t* __restrict__ vtg,
                bf16_t* __restrict__ og, int ldo) {
    __shared__ bf16_t Kt[4][32 * 64];   // [key][d]  (n-major for QK^T B-frags)
    __shared__ bf16_t Vt[4][64 * 32];   // [d][key]  (n-major for PV B-frags)
    __shared__ bf16_t Pt[4][16 * 32];   // [qrow][key]

    const int lane = threadIdx.x & 31;
    const int wave = threadIdx.x >> 5;
    const int bb = blockIdx.y / NH;
    const int hh = blockIdx.y % NH;
    const int q0 = blockIdx.x * 64 + wave * 16;
    const int r  = lane & 15;
    const int hi = lane >> 4;
    const int koffA = hi ? 8 : 0;
    const int koffB = hi ? 8 : 0;

    // Q tile (16x64) held as two 16x32 A-fragments, loaded straight from global
    FragAB aq0, aq1;
    {
        const bf16_t* qrow = qg + (size_t)(bb * TT + q0 + r) * ldqk + hh * HDIM;
        aq0.h[0] = *(const v8bf*)(qrow + koffA);
        aq0.h[1] = *(const v8bf*)(qrow + koffA + 16);
        aq1.h[0] = *(const v8bf*)(qrow + 32 + koffA);
        aq1.h[1] = *(const v8bf*)(qrow + 32 + koffA + 16);
    }
    const bf16_t* vhead = vtg + (size_t)(bb * NH + hh) * HDIM * TT;

    v8f o0 = {}, o1 = {}, o2 = {}, o3 = {};
    float mrow[8], lrow[8];
#pragma unroll
    for (int j = 0; j < 8; ++j) { mrow[j] = -1e30f; lrow[j] = 0.f; }

    const int nblk = (q0 + 16 + 31) >> 5;     // causal bound, 32-key blocks
    for (int kb = 0; kb < nblk; ++kb) {
        const int kbase = kb * 32;
        // stage K: each lane owns one key row (contiguous b128)
        {
            const bf16_t* krow = kg + (size_t)(bb * TT + kbase + lane) * ldqk + hh * HDIM;
#pragma unroll
            for (int c = 0; c < 8; ++c)
                *(v8bf*)&Kt[wave][lane * 64 + c * 8] = *(const v8bf*)(krow + c * 8);
        }
        // stage V^T: each lane owns two d-rows of 32 keys (contiguous b128)
#pragma unroll
        for (int dd = 0; dd < 2; ++dd) {
            const int d = lane + dd * 32;
            const bf16_t* vrow = vhead + (size_t)d * TT + kbase;
#pragma unroll
            for (int c = 0; c < 4; ++c)
                *(v8bf*)&Vt[wave][d * 32 + c * 8] = *(const v8bf*)(vrow + c * 8);
        }
        // intra-wave LDS ops are in-order; no block barrier needed

        // S = Q * K^T : two 16x16 f32 tiles, K-dim 64 = 2 WMMA k-steps each
        v8f s0 = {}, s1 = {};
#pragma unroll
        for (int f = 0; f < 2; ++f) {
            v8f sc = {};
            FragAB bk0, bk1;
            const bf16_t* bp = &Kt[wave][(f * 16 + r) * 64];
            bk0.h[0] = *(const v8bf*)(bp + koffB);
            bk0.h[1] = *(const v8bf*)(bp + koffB + 16);
            bk1.h[0] = *(const v8bf*)(bp + 32 + koffB);
            bk1.h[1] = *(const v8bf*)(bp + 32 + koffB + 16);
            sc = __builtin_amdgcn_wmma_f32_16x16x32_bf16(
                false, aq0.v, false, bk0.v, (short)0, sc, false, false);
            sc = __builtin_amdgcn_wmma_f32_16x16x32_bf16(
                false, aq1.v, false, bk1.v, (short)0, sc, false, false);
            if (f == 0) s0 = sc; else s1 = sc;
        }

        // scale + causal mask + online softmax (row j+8*hi lives across the
        // 16 lanes of this half; reduce with xor shuffles 8..1)
#pragma unroll
        for (int j = 0; j < 8; ++j) {
            const int m    = j + (hi << 3);
            const int qidx = q0 + m;
            float v0 = s0[j] * 0.125f, v1 = s1[j] * 0.125f;
            v0 = (kbase + r      <= qidx) ? v0 : -1e30f;
            v1 = (kbase + 16 + r <= qidx) ? v1 : -1e30f;

            float t = fmaxf(v0, v1);
            for (int msk = 8; msk; msk >>= 1) t = fmaxf(t, __shfl_xor(t, msk, 32));
            const float mnew  = fmaxf(mrow[j], t);
            const float alpha = __expf(mrow[j] - mnew);
            const float p0 = __expf(v0 - mnew);
            const float p1 = __expf(v1 - mnew);
            float rs = p0 + p1;
            for (int msk = 8; msk; msk >>= 1) rs += __shfl_xor(rs, msk, 32);
            lrow[j] = lrow[j] * alpha + rs;
            mrow[j] = mnew;
            o0[j] *= alpha; o1[j] *= alpha; o2[j] *= alpha; o3[j] *= alpha;
            Pt[wave][m * 32 + r]      = (bf16_t)p0;
            Pt[wave][m * 32 + 16 + r] = (bf16_t)p1;
        }

        // O += P (16x32) * V (32x64): 4 WMMAs
        FragAB pa;
        pa.h[0] = *(const v8bf*)&Pt[wave][r * 32 + koffA];
        pa.h[1] = *(const v8bf*)&Pt[wave][r * 32 + koffA + 16];
#pragma unroll
        for (int f = 0; f < 4; ++f) {
            FragAB vb;
            const bf16_t* vp = &Vt[wave][(f * 16 + r) * 32 + koffB];
            vb.h[0] = *(const v8bf*)vp;
            vb.h[1] = *(const v8bf*)(vp + 16);
            v8f* oc = (f == 0) ? &o0 : (f == 1) ? &o1 : (f == 2) ? &o2 : &o3;
            *oc = __builtin_amdgcn_wmma_f32_16x16x32_bf16(
                false, pa.v, false, vb.v, (short)0, *oc, false, false);
        }
    }

    // normalize and write O (bf16, [B*T, D] with head offset)
#pragma unroll
    for (int j = 0; j < 8; ++j) {
        const int m = j + (hi << 3);
        const float inv = 1.0f / lrow[j];
        bf16_t* orow = og + (size_t)(bb * TT + q0 + m) * ldo + hh * HDIM + r;
        orow[0]  = (bf16_t)(o0[j] * inv);
        orow[16] = (bf16_t)(o1[j] * inv);
        orow[32] = (bf16_t)(o2[j] * inv);
        orow[48] = (bf16_t)(o3[j] * inv);
    }
}

// ---------------------------------------------------------------------------
// Host-side orchestration
// ---------------------------------------------------------------------------
extern "C" void kernel_launch(void* const* d_in, const int* in_sizes, int n_in,
                              void* d_out, int out_size, void* d_ws, size_t ws_size,
                              hipStream_t stream) {
    (void)in_sizes; (void)n_in; (void)out_size; (void)ws_size;
    const float* x      = (const float*)d_in[0];
    const float* qk_w   = (const float*)d_in[1];
    const float* qk_b   = (const float*)d_in[2];
    const float* v_w    = (const float*)d_in[3];
    const float* proj_w = (const float*)d_in[4];
    const float* ln1_w  = (const float*)d_in[5];
    const float* ln1_b  = (const float*)d_in[6];
    const float* ln2_w  = (const float*)d_in[7];
    const float* ln2_b  = (const float*)d_in[8];
    const float* fc_w   = (const float*)d_in[9];
    const float* fc_b   = (const float*)d_in[10];
    const float* mlp_w  = (const float*)d_in[11];
    const float* mlp_b  = (const float*)d_in[12];
    float* out = (float*)d_out;

    const int M4 = BB * TT;                 // 4096 rows
    char* ws = (char*)d_ws;
    size_t off = 0;
    auto alloc = [&](size_t bytes) -> char* {
        char* p = ws + off;
        off += (bytes + 255) & ~(size_t)255;
        return p;
    };
    bf16_t* wqkT  = (bf16_t*)alloc((size_t)DDIM * 2 * DDIM * 2);  // [2D][D]
    bf16_t* wvT   = (bf16_t*)alloc((size_t)DDIM * DDIM * 2);     // [D][D]
    bf16_t* wprT  = (bf16_t*)alloc((size_t)DDIM * DDIM * 2);
    bf16_t* wfcT  = (bf16_t*)alloc((size_t)DDIM * FFD * 2);      // [FF][D]
    bf16_t* wmlpT = (bf16_t*)alloc((size_t)FFD * DDIM * 2);      // [D][FF]
    bf16_t* hbf   = (bf16_t*)alloc((size_t)M4 * DDIM * 2);
    bf16_t* qkbf  = (bf16_t*)alloc((size_t)M4 * 2 * DDIM * 2);
    bf16_t* vTbf  = (bf16_t*)alloc((size_t)M4 * DDIM * 2);       // [B,H,HD,T]
    bf16_t* obf   = (bf16_t*)alloc((size_t)M4 * DDIM * 2);
    float*  x1    = (float*) alloc((size_t)M4 * DDIM * 4);
    bf16_t* h2bf  = (bf16_t*)alloc((size_t)M4 * DDIM * 2);
    bf16_t* mbf   = (bf16_t*)alloc((size_t)M4 * FFD * 2);

    // 1) convert + transpose weights to bf16 [N][K]
    cvt_transpose_bf16<<<1024, 256, 0, stream>>>(qk_w,   wqkT,  DDIM, 2 * DDIM);
    cvt_transpose_bf16<<<1024, 256, 0, stream>>>(v_w,    wvT,   DDIM, DDIM);
    cvt_transpose_bf16<<<1024, 256, 0, stream>>>(proj_w, wprT,  DDIM, DDIM);
    cvt_transpose_bf16<<<1024, 256, 0, stream>>>(fc_w,   wfcT,  DDIM, FFD);
    cvt_transpose_bf16<<<1024, 256, 0, stream>>>(mlp_w,  wmlpT, FFD, DDIM);

    // 2) h = LN1(x)  (bf16)
    layernorm_bf16<<<M4, 256, 0, stream>>>(x, ln1_w, ln1_b, hbf, DDIM);

    // 3) qk = h @ qk_w + qk_b (bf16) ; vT = (h @ v_w)^T per-head (bf16)
    gemm_bf16_wmma<true, false, false, true, false>
        <<<dim3((2 * DDIM) / 128, M4 / 128), 256, 0, stream>>>(
            hbf, wqkT, qk_b, nullptr, qkbf, M4, 2 * DDIM, DDIM);
    gemm_bf16_wmma<false, false, false, true, true>
        <<<dim3(DDIM / 128, M4 / 128), 256, 0, stream>>>(
            hbf, wvT, nullptr, nullptr, vTbf, M4, DDIM, DDIM);

    // 4) causal flash attention -> o (bf16)
    flash_attn_wmma<<<dim3(TT / 64, BB * NH), 128, 0, stream>>>(
        qkbf, qkbf + DDIM, 2 * DDIM, vTbf, obf, DDIM);

    // 5) x1 = x + o @ attn_proj_w  (f32 out)
    gemm_bf16_wmma<false, false, true, false, false>
        <<<dim3(DDIM / 128, M4 / 128), 256, 0, stream>>>(
            obf, wprT, nullptr, x, x1, M4, DDIM, DDIM);

    // 6) h2 = LN2(x1) (bf16)
    layernorm_bf16<<<M4, 256, 0, stream>>>(x1, ln2_w, ln2_b, h2bf, DDIM);

    // 7) m = gelu(h2 @ fc_w + fc_b)  (bf16 out)
    gemm_bf16_wmma<true, true, false, true, false>
        <<<dim3(FFD / 128, M4 / 128), 256, 0, stream>>>(
            h2bf, wfcT, fc_b, nullptr, mbf, M4, FFD, DDIM);

    // 8) out = x1 + m @ mlp_proj_w + mlp_proj_b  (f32 out)
    gemm_bf16_wmma<true, false, true, false, false>
        <<<dim3(DDIM / 128, M4 / 128), 256, 0, stream>>>(
            mbf, wmlpT, mlp_b, x1, out, M4, DDIM, FFD);
}
